// AttGraphModel_84430467105381
// MI455X (gfx1250) — matrix operand order repaired and checked by reading, not compiled, not run on hardware
//
#include <hip/hip_runtime.h>

typedef __attribute__((ext_vector_type(16))) __bf16       v16bf;
typedef __attribute__((ext_vector_type(8)))  float        v8f;
typedef __attribute__((ext_vector_type(4)))  unsigned int v4u;

#define HEADS 4
#define NEG_SLOPE 0.2f
#define XS_STRIDE 136   // 128 + 8 pad: keeps 16B alignment of frags, breaks bank conflicts

// ---------- helpers ----------
__device__ __forceinline__ unsigned short f2bf(float f) {
  unsigned u = __float_as_uint(f);
  u += 0x7FFFu + ((u >> 16) & 1u);           // round-to-nearest-even
  return (unsigned short)(u >> 16);
}
// monotonic float <-> uint map so integer atomicMax == float max
__device__ __forceinline__ unsigned encf(float f) {
  unsigned u = __float_as_uint(f);
  return (u & 0x80000000u) ? ~u : (u | 0x80000000u);
}
__device__ __forceinline__ float decf(unsigned u) {
  unsigned v = (u & 0x80000000u) ? (u & 0x7FFFFFFFu) : ~u;
  return __uint_as_float(v);
}
__device__ __forceinline__ float lrelu(float x) { return x > 0.f ? x : NEG_SLOPE * x; }

// ---------- GEMM: feat[N,128] = act(in[N,128]) @ W[128,128], bf16 WMMA ----------
// 4 waves/block, 64 rows/block; each wave: 16 rows x 128 cols = 8 output tiles.
__global__ void __launch_bounds__(128) gemm_bf16_wmma(
    const float* __restrict__ in, const float* __restrict__ W,
    float* __restrict__ feat, int N, int relu_in)
{
  __shared__ unsigned short xs[64  * XS_STRIDE];   // bf16 input tile [row][k]
  __shared__ unsigned short wt[128 * XS_STRIDE];   // bf16 W transposed [n][k]
  const int tid  = threadIdx.x;
  const int row0 = blockIdx.x * 64;

  for (int i = 0; i < 64; ++i) {                   // stage X tile (coalesced)
    int r = row0 + i;
    float v = (r < N) ? in[r * 128 + tid] : 0.f;
    if (relu_in) v = fmaxf(v, 0.f);
    xs[i * XS_STRIDE + tid] = f2bf(v);
  }
  for (int i = 0; i < 128; ++i)                    // stage W^T (coalesced read)
    wt[tid * XS_STRIDE + i] = f2bf(W[i * 128 + tid]);
  __syncthreads();

  const int wave = tid >> 5;
  const int lane = tid & 31;
  const int half = lane >> 4;                      // lane half per ISA frag layout
  const int l16  = lane & 15;

  v8f acc[8];
#pragma unroll
  for (int t = 0; t < 8; ++t) acc[t] = (v8f){0,0,0,0,0,0,0,0};

#pragma unroll
  for (int kt = 0; kt < 4; ++kt) {
    // A frag: row = wave*16 + l16 ; 16 contiguous K at kt*32 + half*16
    const unsigned short* ap = &xs[(wave * 16 + l16) * XS_STRIDE + kt * 32 + half * 16];
    union { v4u q[2]; v16bf v; } ua;
    ua.q[0] = *(const v4u*)(ap);
    ua.q[1] = *(const v4u*)(ap + 8);
    v16bf a = ua.v;
#pragma unroll
    for (int nt = 0; nt < 8; ++nt) {
      // B frag: col = nt*16 + l16 ; 16 contiguous K at kt*32 + half*16 (W^T layout)
      const unsigned short* bp = &wt[(nt * 16 + l16) * XS_STRIDE + kt * 32 + half * 16];
      union { v4u q[2]; v16bf v; } ub;
      ub.q[0] = *(const v4u*)(bp);
      ub.q[1] = *(const v4u*)(bp + 8);
      acc[nt] = __builtin_amdgcn_wmma_f32_16x16x32_bf16(
          false, a, false, ub.v, (short)0, acc[nt], false, false);
    }
  }
  // C layout: VGPR r, lanes 0-15 -> M=r, lanes 16-31 -> M=8+r; N = l16
  const int mrow = row0 + wave * 16 + half * 8;
#pragma unroll
  for (int nt = 0; nt < 8; ++nt)
#pragma unroll
    for (int r = 0; r < 8; ++r) {
      int row = mrow + r;
      if (row < N) feat[row * 128 + nt * 16 + l16] = acc[nt][r];
    }
}

// ---------- el/er per node: wave-per-node dot with a_l/a_r ----------
__global__ void node_attn(const float* __restrict__ feat,
                          const float* __restrict__ al, const float* __restrict__ ar,
                          float* __restrict__ el, float* __restrict__ er, int N)
{
  int g = blockIdx.x * blockDim.x + threadIdx.x;
  int node = g >> 5;
  int lane = g & 31;
  if (node >= N) return;                            // uniform per wave
  float sl[HEADS], sr[HEADS];
#pragma unroll
  for (int h = 0; h < HEADS; ++h) {
    float f = feat[node * 128 + h * 32 + lane];
    sl[h] = f * al[h * 32 + lane];
    sr[h] = f * ar[h * 32 + lane];
  }
#pragma unroll
  for (int off = 16; off; off >>= 1)
#pragma unroll
    for (int h = 0; h < HEADS; ++h) {
      sl[h] += __shfl_xor(sl[h], off, 32);
      sr[h] += __shfl_xor(sr[h], off, 32);
    }
  if (lane == 0)
#pragma unroll
    for (int h = 0; h < HEADS; ++h) {
      el[node * 4 + h] = sl[h];
      er[node * 4 + h] = sr[h];
    }
}

__global__ void fill_u32(unsigned* __restrict__ p, unsigned val, long long n) {
  long long i = (long long)blockIdx.x * blockDim.x + threadIdx.x;
  if (i < n) p[i] = val;
}

// ---------- edge pass 1: logits + segment max ----------
__global__ void edge_logit_max(const int* __restrict__ src, const int* __restrict__ dst,
                               const float* __restrict__ el, const float* __restrict__ er,
                               float* __restrict__ e, unsigned* __restrict__ nmax, int E)
{
  int i = blockIdx.x * blockDim.x + threadIdx.x;
  if (i >= E) return;
  int s = src[i], d = dst[i];
  float4 a = *(const float4*)(el + 4 * s);
  float4 b = *(const float4*)(er + 4 * d);
  float4 v = make_float4(lrelu(a.x + b.x), lrelu(a.y + b.y),
                         lrelu(a.z + b.z), lrelu(a.w + b.w));
  *(float4*)(e + 4 * i) = v;
  atomicMax(&nmax[d * 4 + 0], encf(v.x));
  atomicMax(&nmax[d * 4 + 1], encf(v.y));
  atomicMax(&nmax[d * 4 + 2], encf(v.z));
  atomicMax(&nmax[d * 4 + 3], encf(v.w));
}

// ---------- edge pass 2: exp + segment sum (overwrites e with ee) ----------
__global__ void edge_exp_sum(const int* __restrict__ dst, float* __restrict__ e,
                             const unsigned* __restrict__ nmax,
                             float* __restrict__ denom, int E)
{
  int i = blockIdx.x * blockDim.x + threadIdx.x;
  if (i >= E) return;
  int d = dst[i];
  float4 v = *(const float4*)(e + 4 * i);
  v.x = __expf(v.x - decf(nmax[d * 4 + 0]));
  v.y = __expf(v.y - decf(nmax[d * 4 + 1]));
  v.z = __expf(v.z - decf(nmax[d * 4 + 2]));
  v.w = __expf(v.w - decf(nmax[d * 4 + 3]));
  *(float4*)(e + 4 * i) = v;
  atomicAdd(&denom[d * 4 + 0], v.x);
  atomicAdd(&denom[d * 4 + 1], v.y);
  atomicAdd(&denom[d * 4 + 2], v.z);
  atomicAdd(&denom[d * 4 + 3], v.w);
}

// ---------- edge pass 3: weighted scatter-add, 32 lanes per edge ----------
__global__ void edge_scatter(const int* __restrict__ src, const int* __restrict__ dst,
                             const float* __restrict__ ee, const float* __restrict__ denom,
                             const float* __restrict__ feat, float* __restrict__ agg, int E)
{
  long long t = (long long)blockIdx.x * blockDim.x + threadIdx.x;
  int e = (int)(t >> 5);
  if (e >= E) return;                               // uniform per wave
  int lane = (int)(t & 31);
  int s = src[e], d = dst[e];
  int h = lane >> 3;                                // 8 lanes (32 cols) per head
  float alpha = ee[e * 4 + h] / denom[d * 4 + h];
  float4 f = *(const float4*)(feat + s * 128 + lane * 4);
  float* o = agg + d * 128 + lane * 4;
  atomicAdd(o + 0, f.x * alpha);
  atomicAdd(o + 1, f.y * alpha);
  atomicAdd(o + 2, f.z * alpha);
  atomicAdd(o + 3, f.w * alpha);
}

// ---------- relu + graph max-pool over nodes ----------
__global__ void relu_maxpool(const float* __restrict__ h, unsigned* __restrict__ pooled, int N)
{
  int col = threadIdx.x;                            // blockDim = 128
  float m = 0.f;                                    // init 0 == fused relu
  for (int r = blockIdx.x; r < N; r += gridDim.x)
    m = fmaxf(m, h[r * 128 + col]);
  atomicMax(&pooled[col], encf(m));
}

// ---------- FC + softmax readout ----------
__global__ void readout(const unsigned* __restrict__ pooled,
                        const float* __restrict__ Wfc, const float* __restrict__ bfc,
                        float* __restrict__ out)
{
  __shared__ float p[128];
  int t = threadIdx.x;
  if (t < 128) p[t] = decf(pooled[t]);
  __syncthreads();
  if (t == 0) {
    float logit[8]; float mx = -1e30f;
    for (int j = 0; j < 8; ++j) {
      float s = bfc[j];
      for (int k = 0; k < 128; ++k) s += p[k] * Wfc[k * 8 + j];
      logit[j] = s; mx = fmaxf(mx, s);
    }
    float den = 0.f;
    for (int j = 0; j < 8; ++j) { logit[j] = __expf(logit[j] - mx); den += logit[j]; }
    for (int j = 0; j < 8; ++j) out[j] = logit[j] / den;
  }
}

extern "C" void kernel_launch(void* const* d_in, const int* in_sizes, int n_in,
                              void* d_out, int out_size, void* d_ws, size_t ws_size,
                              hipStream_t stream) {
  const float* x   = (const float*)d_in[0];
  const int*   src = (const int*)  d_in[1];
  const int*   dst = (const int*)  d_in[2];
  const float* Wl[3]  = { (const float*)d_in[3], (const float*)d_in[6], (const float*)d_in[9]  };
  const float* all[3] = { (const float*)d_in[4], (const float*)d_in[7], (const float*)d_in[10] };
  const float* arl[3] = { (const float*)d_in[5], (const float*)d_in[8], (const float*)d_in[11] };
  const float* Wfc = (const float*)d_in[12];
  const float* bfc = (const float*)d_in[13];

  const int N = in_sizes[0] / 128;
  const int E = in_sizes[1];

  char* ws = (char*)d_ws;
  size_t off = 0;
  auto take = [&](size_t bytes) -> void* {
    void* p = ws + off;
    off += (bytes + 255) & ~(size_t)255;
    return p;
  };
  float*    featA  = (float*)   take((size_t)N * 128 * 4);
  float*    featB  = (float*)   take((size_t)N * 128 * 4);
  float*    ebuf   = (float*)   take((size_t)E * 4 * 4);
  float*    el     = (float*)   take((size_t)N * 4 * 4);
  float*    er     = (float*)   take((size_t)N * 4 * 4);
  unsigned* nmax   = (unsigned*)take((size_t)N * 4 * 4);
  float*    denom  = (float*)   take((size_t)N * 4 * 4);
  unsigned* pooled = (unsigned*)take(128 * 4);
  (void)ws_size; (void)n_in; (void)out_size;

  const int gGemm  = (N + 63) / 64;
  const int gNode  = (N * 32 + 255) / 256;
  const int gNH    = (N * 4 + 255) / 256;
  const int gAgg   = (int)(((long long)N * 128 + 255) / 256);
  const int gEdge  = (E + 255) / 256;
  const int gScat  = (int)(((long long)E * 32 + 255) / 256);

  auto gat = [&](const float* inp, int li, float* feat, float* agg, int relu_in) {
    gemm_bf16_wmma<<<gGemm, 128, 0, stream>>>(inp, Wl[li], feat, N, relu_in);
    node_attn<<<gNode, 256, 0, stream>>>(feat, all[li], arl[li], el, er, N);
    fill_u32<<<gNH, 256, 0, stream>>>(nmax, 0x007FFFFFu /*enc(-inf)*/, (long long)N * 4);
    fill_u32<<<gNH, 256, 0, stream>>>((unsigned*)denom, 0u, (long long)N * 4);
    fill_u32<<<gAgg, 256, 0, stream>>>((unsigned*)agg, 0u, (long long)N * 128);
    edge_logit_max<<<gEdge, 256, 0, stream>>>(src, dst, el, er, ebuf, nmax, E);
    edge_exp_sum<<<gEdge, 256, 0, stream>>>(dst, ebuf, nmax, denom, E);
    edge_scatter<<<gScat, 256, 0, stream>>>(src, dst, ebuf, denom, feat, agg, E);
  };

  gat(x,     0, featA, featB, 0);   // layer 1: relu applied at next stage load
  gat(featB, 1, featA, featB, 1);   // layer 2
  gat(featB, 2, featA, featB, 1);   // layer 3 (no activation; final relu in pool)

  fill_u32<<<1, 128, 0, stream>>>(pooled, 0x80000000u /*enc(0)*/, 128);
  relu_maxpool<<<768, 128, 0, stream>>>(featB, pooled, N);
  readout<<<1, 128, 0, stream>>>(pooled, Wfc, bfc, (float*)d_out);
}
